// AttentionMatcher_14027363189464
// MI455X (gfx1250) — compile-verified, matrix operand order checked by reading, not run on hardware
//
#include <hip/hip_runtime.h>
#include <cstdint>

#define NB    16
#define LQ    2048
#define LS    2048
#define DD    512
#define KSTEPS (DD / 32)           // 16 wmma k-steps
#define STILE  16                  // s-rows per tile
#define NTILES (LS / STILE)        // 128
#define LDS_STRIDE 520             // bf16 elems per LDS row (512 + 8 pad) -> conflict-free
#define TILE_ELEMS (STILE * LDS_STRIDE)
#define TILE_BYTES (TILE_ELEMS * 2)
#define LOG2E 1.4426950408889634f
#define LN2   0.69314718055994531f

typedef __attribute__((ext_vector_type(16))) __bf16 v16bf;
typedef __attribute__((ext_vector_type(8)))  __bf16 v8bf;
typedef __attribute__((ext_vector_type(8)))  float  v8f;

__device__ __forceinline__ float fexp2(float x) { return __builtin_amdgcn_exp2f(x); }

// ---------------- fp32 -> bf16 conversion (optionally pre-scaled) ----------------
__global__ __launch_bounds__(256) void cvt_bf16_kernel(const float* __restrict__ in,
                                                       __bf16* __restrict__ out,
                                                       float scale, int n8) {
  int i = blockIdx.x * blockDim.x + threadIdx.x;
  if (i >= n8) return;
  const float4* p = (const float4*)(in + (size_t)i * 8);
  float4 a = p[0];
  float4 c = p[1];
  v8bf o;
  o[0] = (__bf16)(a.x * scale); o[1] = (__bf16)(a.y * scale);
  o[2] = (__bf16)(a.z * scale); o[3] = (__bf16)(a.w * scale);
  o[4] = (__bf16)(c.x * scale); o[5] = (__bf16)(c.y * scale);
  o[6] = (__bf16)(c.z * scale); o[7] = (__bf16)(c.w * scale);
  *(v8bf*)(out + (size_t)i * 8) = o;
}

// ---------------- fused QK^T + softmax-expectation kernel ----------------
// grid: NB * (LQ/128) blocks, 256 threads (8 waves); wave w owns q rows [qb0+16w, +16)
__global__ __launch_bounds__(256) void attn_score_kernel(const __bf16* __restrict__ Qbf,
                                                         const __bf16* __restrict__ Sbf,
                                                         float* __restrict__ out) {
  __shared__ __align__(16) __bf16 smem[3 * TILE_ELEMS];   // triple-buffered S tiles

  const int tid  = threadIdx.x;
  const int lane = tid & 31;
  const int wave = tid >> 5;       // 0..7
  const int h    = lane >> 4;      // K-half selector (WMMA operand layout)
  const int ln   = lane & 15;      // row (A) / col (B,C) index
  const int b    = blockIdx.x >> 4;
  const int qb0  = (blockIdx.x & 15) * 128;

  const uint64_t sbase = (uint64_t)(uintptr_t)(Sbf + (size_t)b * LS * DD);
  const uint32_t lds0  = (uint32_t)(uintptr_t)(&smem[0]);

  // per-thread chunk coordinates for async tile copies (1024 x 16B chunks / tile)
  uint32_t crow[4], ccol[4];
#pragma unroll
  for (int i = 0; i < 4; ++i) {
    uint32_t idx = (uint32_t)(tid + i * 256);
    crow[i] = idx >> 6;
    ccol[i] = (idx & 63) * 16;
  }

  // ---- A fragments: this wave's 16 q rows x full D=512, resident in VGPRs ----
  const __bf16* qrow = Qbf + ((size_t)b * LQ + qb0 + wave * 16 + ln) * DD;
  v16bf afrag[KSTEPS];
#pragma unroll
  for (int kk = 0; kk < KSTEPS; ++kk) {
    v8bf lo = *(const v8bf*)(qrow + kk * 32 + h * 8);
    v8bf hi = *(const v8bf*)(qrow + kk * 32 + 16 + h * 8);
    afrag[kk] = __builtin_shufflevector(lo, hi, 0, 1, 2, 3, 4, 5, 6, 7,
                                        8, 9, 10, 11, 12, 13, 14, 15);
  }

  // per-lane online softmax-expectation state (8 C-rows per lane), exp2 domain
  float mrun[8], Zrun[8], Wrun[8];
#pragma unroll
  for (int r = 0; r < 8; ++r) { mrun[r] = -3.0e38f; Zrun[r] = 0.f; Wrun[r] = 0.f; }

  // ---- prologue: async-copy S tiles 0 and 1 into LDS buffers 0 and 1 ----
#pragma unroll
  for (int tt = 0; tt < 2; ++tt) {
    const uint32_t ldsb  = lds0 + (uint32_t)tt * TILE_BYTES;
    const uint32_t gtile = (uint32_t)tt * (STILE * DD * 2);
#pragma unroll
    for (int i = 0; i < 4; ++i) {
      uint32_t lds_addr = ldsb + crow[i] * (LDS_STRIDE * 2) + ccol[i];
      uint32_t goff = gtile + crow[i] * (DD * 2) + ccol[i];
      asm volatile("global_load_async_to_lds_b128 %0, %1, %2 offset:0"
                   :: "v"(lds_addr), "v"(goff), "s"(sbase) : "memory");
    }
  }

  int p = 0;                         // t % 3 (buffer index), tracked incrementally
  for (int t = 0; t < NTILES; ++t) {
    // tile t's 4 async ops complete (ordered) once <=4 remain (tile t+1 in flight)
    if (t == NTILES - 1) asm volatile("s_wait_asynccnt 0" ::: "memory");
    else                 asm volatile("s_wait_asynccnt 4" ::: "memory");
    __syncthreads();   // all waves' tile-t data resident; also fences buf reuse

    // issue tile t+2 into buffer (t+2)%3: every wave passed the barrier above,
    // hence finished reading tile t-1 from that buffer (distance-3 reuse)
    if (t + 2 < NTILES) {
      int pn = p - 1; if (pn < 0) pn += 3;           // (t+2)%3
      const uint32_t ldsb  = lds0 + (uint32_t)pn * TILE_BYTES;
      const uint32_t gtile = (uint32_t)(t + 2) * (STILE * DD * 2);
#pragma unroll
      for (int i = 0; i < 4; ++i) {
        uint32_t lds_addr = ldsb + crow[i] * (LDS_STRIDE * 2) + ccol[i];
        uint32_t goff = gtile + crow[i] * (DD * 2) + ccol[i];
        asm volatile("global_load_async_to_lds_b128 %0, %1, %2 offset:0"
                     :: "v"(lds_addr), "v"(goff), "s"(sbase) : "memory");
      }
    }

    // ---- 16x16 logits tile via 16 bf16 WMMAs over K=512 ----
    // B fragments: explicit 3-slot ring, prefetch distance 2 -> ds_loads stay
    // 4-deep in flight behind each WMMA instead of a full s_wait_dscnt 0.
    const __bf16* srow = smem + p * TILE_ELEMS + ln * LDS_STRIDE;
    v8f acc = {0.f, 0.f, 0.f, 0.f, 0.f, 0.f, 0.f, 0.f};
    v8bf blo[3], bhi[3];
#pragma unroll
    for (int kk = 0; kk < 2; ++kk) {
      blo[kk] = *(const v8bf*)(srow + kk * 32 + h * 8);
      bhi[kk] = *(const v8bf*)(srow + kk * 32 + 16 + h * 8);
    }
#pragma unroll
    for (int kk = 0; kk < KSTEPS; ++kk) {
      if (kk + 2 < KSTEPS) {
        blo[(kk + 2) % 3] = *(const v8bf*)(srow + (kk + 2) * 32 + h * 8);
        bhi[(kk + 2) % 3] = *(const v8bf*)(srow + (kk + 2) * 32 + 16 + h * 8);
      }
      v16bf bfrag = __builtin_shufflevector(blo[kk % 3], bhi[kk % 3],
                                            0, 1, 2, 3, 4, 5, 6, 7,
                                            8, 9, 10, 11, 12, 13, 14, 15);
      acc = __builtin_amdgcn_wmma_f32_16x16x32_bf16(
          false, afrag[kk], false, bfrag, (short)0, acc, false, false);
    }

    // ---- per-lane streaming log-sum-exp expectation update (no shuffles) ----
#pragma unroll
    for (int r = 0; r < 8; ++r) {
      float tv = acc[r];                     // log2-domain logit
      float nm = fmaxf(mrun[r], tv);
      float e1 = fexp2(mrun[r] - nm);
      float e2 = fexp2(tv - nm);
      Zrun[r] = Zrun[r] * e1 + e2;
      Wrun[r] = Wrun[r] * e1 + e2 * tv;
      mrun[r] = nm;
    }

    ++p; if (p == 3) p = 0;
  }

  // ---- merge 16 lane-partials per C-row (width-16 xor tree), write scores ----
#pragma unroll
  for (int r = 0; r < 8; ++r) {
    float m = mrun[r], Z = Zrun[r], W = Wrun[r];
#pragma unroll
    for (int mask = 1; mask <= 8; mask <<= 1) {
      float om = __shfl_xor(m, mask, 16);
      float oZ = __shfl_xor(Z, mask, 16);
      float oW = __shfl_xor(W, mask, 16);
      float nm = fmaxf(m, om);
      float e1 = fexp2(m - nm);
      float e2 = fexp2(om - nm);
      Z = Z * e1 + oZ * e2;
      W = W * e1 + oW * e2;
      m = nm;
    }
    if (ln == 0) {
      // row M = 8*h + r of this wave's 16-row band; convert exp2-domain back
      out[(size_t)b * LQ + qb0 + wave * 16 + h * 8 + r] = LN2 * (W / Z);
    }
  }
}

// ---------------- host launcher ----------------
extern "C" void kernel_launch(void* const* d_in, const int* in_sizes, int n_in,
                              void* d_out, int out_size, void* d_ws, size_t ws_size,
                              hipStream_t stream) {
  (void)in_sizes; (void)n_in; (void)out_size; (void)ws_size;
  const float* Q = (const float*)d_in[0];   // [16, 2048, 512] f32
  const float* S = (const float*)d_in[1];   // [16, 2048, 512] f32
  float* out = (float*)d_out;               // [16, 2048] f32

  __bf16* Qbf = (__bf16*)d_ws;                          // 32 MB
  __bf16* Sbf = Qbf + (size_t)NB * LQ * DD;             // +32 MB

  const int nQ8 = NB * LQ * DD / 8;  // 2097152
  const int nS8 = NB * LS * DD / 8;
  // Q pre-scaled by log2(e): GEMM emits logits directly in exp2 domain
  cvt_bf16_kernel<<<(nQ8 + 255) / 256, 256, 0, stream>>>(Q, Qbf, LOG2E, nQ8);
  cvt_bf16_kernel<<<(nS8 + 255) / 256, 256, 0, stream>>>(S, Sbf, 1.0f, nS8);

  attn_score_kernel<<<NB * (LQ / 128), 256, 0, stream>>>(Qbf, Sbf, out);
}
